// EOPA_8306466751030
// MI455X (gfx1250) — matrix operand order repaired and checked by reading, not compiled, not run on hardware
//
#include <hip/hip_runtime.h>
#include <hip/hip_bf16.h>

typedef unsigned short u16;
typedef unsigned int   u32;

typedef __attribute__((ext_vector_type(16))) __bf16 v16bf;
typedef __attribute__((ext_vector_type(8)))  __bf16 v8bf;
typedef __attribute__((ext_vector_type(8)))  float  v8f;

union BFrag { v16bf v; v8bf h[2]; };

#define D 128
#define G 384            // 3*D gate width
#define WAVES 8          // waves per block
#define TILE_M 16        // nodes per wave
// ws byte offsets
#define WS_STAT   0          // sum[128], sq[128], scale[128], shift[128]
#define WS_WIH    4096       // 384*128 bf16 = 98304 B (W_hh follows contiguously)
#define WS_WHH    102400
#define WS_WSELF  200704     // 128*128 bf16 = 32768 B
#define WS_WNEIGH 233472
#define WS_XBF    266240     // N*128 bf16
// LDS layout (dynamic)
#define LDS_W_BYTES   196608                 // W_ih + W_hh bf16
#define LDS_H_BYTES   (WAVES * TILE_M * D * 2)
#define LDS_TOTAL     (LDS_W_BYTES + LDS_H_BYTES)

static __device__ __forceinline__ u16 f2bf(float f) {
    u32 u = __float_as_uint(f);
    u = (u + 0x7FFFu + ((u >> 16) & 1u)) >> 16;
    return (u16)u;
}
static __device__ __forceinline__ float bf2f(u16 b) {
    return __uint_as_float(((u32)b) << 16);
}
static __device__ __forceinline__ float fsig(float x) {
    return 1.0f / (1.0f + __expf(-x));
}
static __device__ __forceinline__ float ftanh(float x) {
    return 1.0f - 2.0f / (__expf(2.0f * x) + 1.0f);
}
static __device__ __forceinline__ v8f splat8(float b) {
    v8f c;
#pragma unroll
    for (int i = 0; i < 8; ++i) c[i] = b;
    return c;
}
static __device__ __forceinline__ v8f wmma_bf(const BFrag& a, const BFrag& b, v8f c) {
    return __builtin_amdgcn_wmma_f32_16x16x32_bf16(
        false, a.v, false, b.v, (short)0, c, false, false);
}
// A-matrix 16x32 bf16 fragment: lane m = l&15, half hh = l>>4.
// element e<8 -> K = 32*kt + e + 8*hh ; e>=8 -> K = 32*kt + 16 + (e-8) + 8*hh
static __device__ __forceinline__ void loadA(BFrag& f, const u16* row, int kt, int hh) {
    const u16* p = row + kt * 32 + 8 * hh;
    f.h[0] = *(const v8bf*)(p);
    f.h[1] = *(const v8bf*)(p + 16);
}
// B-matrix 32x16 bf16 fragment: lane n = l&15, half hh = l>>4.
// element e<8 -> K = 32*kt + e + 8*hh ; e>=8 -> K = 32*kt + 16 + (e-8) + 8*hh (row n of W)
static __device__ __forceinline__ void loadB(BFrag& f, const u16* wrow, int kt, int hh) {
    const u16* p = wrow + kt * 32 + 8 * hh;
    f.h[0] = *(const v8bf*)(p);
    f.h[1] = *(const v8bf*)(p + 16);
}

// ---------------- stats / BN prep kernels ----------------
__global__ void k_zero_stats(float* stat) {
    if (threadIdx.x < 256) stat[threadIdx.x] = 0.0f;
}

__global__ void k_bn_stats(const float* __restrict__ feat, float* __restrict__ stat, int N) {
    __shared__ float s_sum[256];
    __shared__ float s_sq[256];
    int d = threadIdx.x & 127;
    int half = threadIdx.x >> 7;
    float s = 0.0f, q = 0.0f;
    for (long r = (long)blockIdx.x * 2 + half; r < N; r += (long)gridDim.x * 2) {
        float v = feat[r * D + d];
        s += v; q += v * v;
    }
    s_sum[threadIdx.x] = s;
    s_sq[threadIdx.x]  = q;
    __syncthreads();
    if (half == 0) {
        atomicAdd(&stat[d],       s_sum[d] + s_sum[d + 128]);
        atomicAdd(&stat[128 + d], s_sq[d]  + s_sq[d + 128]);
    }
}

__global__ void k_bn_finalize(float* stat, const float* __restrict__ gamma,
                              const float* __restrict__ beta, int N) {
    int d = threadIdx.x;
    if (d < 128) {
        float inv_n = 1.0f / (float)N;
        float mu  = stat[d] * inv_n;
        float var = stat[128 + d] * inv_n - mu * mu;
        float sc  = gamma[d] * __frsqrt_rn(var + 1e-5f);
        stat[256 + d] = sc;               // scale
        stat[384 + d] = beta[d] - mu * sc;// shift
    }
}

__global__ void k_conv_weights(const float* __restrict__ Wih, const float* __restrict__ Whh,
                               const float* __restrict__ Wself, const float* __restrict__ Wneigh,
                               u16* __restrict__ wihb, u16* __restrict__ whhb,
                               u16* __restrict__ wselfb, u16* __restrict__ wneighb) {
    int i = blockIdx.x * blockDim.x + threadIdx.x;   // 0 .. 131071
    if (i < 49152)            wihb[i]              = f2bf(Wih[i]);
    else if (i < 98304)       whhb[i - 49152]      = f2bf(Whh[i - 49152]);
    else if (i < 114688)      wselfb[i - 98304]    = f2bf(Wself[i - 98304]);
    else if (i < 131072)      wneighb[i - 114688]  = f2bf(Wneigh[i - 114688]);
}

__global__ void k_normalize(const float* __restrict__ feat, const float* __restrict__ stat,
                            u16* __restrict__ xbf, int N) {
    long g = (long)blockIdx.x * blockDim.x + threadIdx.x;   // group of 4 elems
    long total = (long)N * D / 4;
    if (g >= total) return;
    long e = g * 4;
    int d = (int)(e & (D - 1));
    const float4 v = *(const float4*)(feat + e);
    float4 sc = *(const float4*)(stat + 256 + d);
    float4 sh = *(const float4*)(stat + 384 + d);
    u32 lo = (u32)f2bf(v.x * sc.x + sh.x) | ((u32)f2bf(v.y * sc.y + sh.y) << 16);
    u32 hi = (u32)f2bf(v.z * sc.z + sh.z) | ((u32)f2bf(v.w * sc.w + sh.w) << 16);
    uint2 pk; pk.x = lo; pk.y = hi;
    *(uint2*)(xbf + e) = pk;
}

// ---------------- fused ragged-GRU + output projection ----------------
__global__ void __launch_bounds__(256, 1)
k_eopa_gru(const u16* __restrict__ xbf,      // [N,128] bf16 (L2-resident)
           const u16* __restrict__ wboth,    // W_ih then W_hh bf16, contiguous
           const u16* __restrict__ wselfb,   // [128,128] bf16
           const u16* __restrict__ wneighb,  // [128,128] bf16
           const float* __restrict__ bih,    // [384]
           const float* __restrict__ bhh,    // [384]
           const int* __restrict__ srcp,     // [N,MAXD]
           const unsigned char* __restrict__ mskp, // [N,MAXD] bool
           float* __restrict__ out,          // [N,128]
           int N, int MAXD) {
    extern __shared__ char smem[];
    u16* lds_w = (u16*)smem;                         // W_ih(49152) + W_hh(49152)
    const int tid  = threadIdx.x;
    const int wave = tid >> 5;
    const int lane = tid & 31;
    const int jm   = lane & 15;   // node row within tile (A layout) / column n (B,C,D layout)
    const int hh   = lane >> 4;   // lane half

    // cooperative stage of both weight matrices into LDS (12288 x 16B)
    {
        const uint4* gsrc = (const uint4*)wboth;
        uint4* ldst = (uint4*)lds_w;
#pragma unroll 1
        for (int i = tid; i < (LDS_W_BYTES / 16); i += 256) ldst[i] = gsrc[i];
    }
    // per-wave h tile, zero init (256 x 16B per wave)
    u16* hbuf = (u16*)(smem + LDS_W_BYTES) + wave * (TILE_M * D);
    {
        uint4* hz = (uint4*)hbuf;
        uint4 z; z.x = z.y = z.z = z.w = 0u;
#pragma unroll
        for (int i = lane; i < (TILE_M * D * 2 / 16); i += 32) hz[i] = z;
    }
    __syncthreads();

    const int base = (blockIdx.x * WAVES + wave) * TILE_M;
    if (base >= N) return;

    const u16* lds_wih = lds_w;
    const u16* lds_whh = lds_w + 49152;
    const u16* hrow = hbuf + jm * D;

    // rotating one-step lookahead of src/mask (per lane: node base+jm)
    const long row0 = (long)(base + jm) * MAXD;
    int   src_next = srcp[row0];
    float mk_next  = mskp[row0] ? 1.0f : 0.0f;
    __builtin_prefetch(xbf + (long)src_next * D, 0, 1);
    __builtin_prefetch(xbf + (long)src_next * D + 64, 0, 1);

#pragma unroll 1
    for (int t = 0; t < MAXD; ++t) {
        const int   src = src_next;
        const float mk  = mk_next;
        if (t + 1 < MAXD) {
            src_next = srcp[row0 + t + 1];
            mk_next  = mskp[row0 + t + 1] ? 1.0f : 0.0f;
            // prefetch next step's gather row (256B) into caches while we do WMMAs
            __builtin_prefetch(xbf + (long)src_next * D, 0, 1);
            __builtin_prefetch(xbf + (long)src_next * D + 64, 0, 1);
        }
        float mf[8];
#pragma unroll
        for (int v = 0; v < 8; ++v) mf[v] = __shfl(mk, v + 8 * hh, 32);

        // A fragments: gathered neighbor row (xt) and current h
        BFrag axt[4], ah[4];
        const u16* xrow = xbf + (long)src * D;
#pragma unroll
        for (int kt = 0; kt < 4; ++kt) {
            loadA(axt[kt], xrow, kt, hh);
            loadA(ah[kt],  hrow, kt, hh);
        }

#pragma unroll 1
        for (int ft = 0; ft < 8; ++ft) {
            const int n = ft * 16 + jm;            // output-feature column
            v8f a_ir = splat8(bih[n]);
            v8f a_iz = splat8(bih[128 + n]);
            v8f a_in = splat8(bih[256 + n]);
            v8f a_hr = splat8(bhh[n]);
            v8f a_hz = splat8(bhh[128 + n]);
            v8f a_hn = splat8(bhh[256 + n]);
            const u16* wi0 = lds_wih + (long)n * D;
            const u16* wh0 = lds_whh + (long)n * D;
            BFrag b;
#pragma unroll
            for (int kt = 0; kt < 4; ++kt) {
                loadB(b, wi0,               kt, hh); a_ir = wmma_bf(axt[kt], b, a_ir);
                loadB(b, wi0 + 128 * D,     kt, hh); a_iz = wmma_bf(axt[kt], b, a_iz);
                loadB(b, wi0 + 256 * D,     kt, hh); a_in = wmma_bf(axt[kt], b, a_in);
                loadB(b, wh0,               kt, hh); a_hr = wmma_bf(ah[kt],  b, a_hr);
                loadB(b, wh0 + 128 * D,     kt, hh); a_hz = wmma_bf(ah[kt],  b, a_hz);
                loadB(b, wh0 + 256 * D,     kt, hh); a_hn = wmma_bf(ah[kt],  b, a_hn);
            }
            // GRU elementwise update on the C/D layout: element (m = v + 8*hh, n)
#pragma unroll
            for (int v = 0; v < 8; ++v) {
                float r  = fsig(a_ir[v] + a_hr[v]);
                float z  = fsig(a_iz[v] + a_hz[v]);
                float ng = ftanh(a_in[v] + r * a_hn[v]);
                int hidx = (v + 8 * hh) * D + n;
                float ho = bf2f(hbuf[hidx]);
                float hv = (1.0f - z) * ng + z * ho;
                hbuf[hidx] = f2bf(ho + mf[v] * (hv - ho));
            }
        }
    }

    // final: out = x @ W_self^T + h @ W_neigh^T
    BFrag axs[4], ahf[4];
    const u16* xrow = xbf + (long)(base + jm) * D;
#pragma unroll
    for (int kt = 0; kt < 4; ++kt) {
        loadA(axs[kt], xrow, kt, hh);
        loadA(ahf[kt], hrow, kt, hh);
    }
#pragma unroll 1
    for (int ft = 0; ft < 8; ++ft) {
        const int n = ft * 16 + jm;
        v8f acc = splat8(0.0f);
        const u16* ws0 = wselfb  + (long)n * D;
        const u16* wn0 = wneighb + (long)n * D;
        BFrag b;
#pragma unroll
        for (int kt = 0; kt < 4; ++kt) {
            loadB(b, ws0, kt, hh); acc = wmma_bf(axs[kt], b, acc);
            loadB(b, wn0, kt, hh); acc = wmma_bf(ahf[kt], b, acc);
        }
#pragma unroll
        for (int v = 0; v < 8; ++v) {
            out[(long)(base + v + 8 * hh) * D + n] = acc[v];
        }
    }
}

extern "C" void kernel_launch(void* const* d_in, const int* in_sizes, int n_in,
                              void* d_out, int out_size, void* d_ws, size_t ws_size,
                              hipStream_t stream) {
    const float* feat   = (const float*)d_in[0];
    const float* gamma  = (const float*)d_in[1];
    const float* beta   = (const float*)d_in[2];
    const float* Wih    = (const float*)d_in[3];
    const float* Whh    = (const float*)d_in[4];
    const float* bih    = (const float*)d_in[5];
    const float* bhh    = (const float*)d_in[6];
    const float* Wself  = (const float*)d_in[7];
    const float* Wneigh = (const float*)d_in[8];
    const int*   srcp   = (const int*)d_in[9];
    const unsigned char* mskp = (const unsigned char*)d_in[10];
    float* out = (float*)d_out;

    const int N    = in_sizes[0] / D;
    const int MAXD = in_sizes[9] / N;

    char* ws = (char*)d_ws;
    float* stat   = (float*)(ws + WS_STAT);
    u16* wihb     = (u16*)(ws + WS_WIH);
    u16* whhb     = (u16*)(ws + WS_WHH);
    u16* wselfb   = (u16*)(ws + WS_WSELF);
    u16* wneighb  = (u16*)(ws + WS_WNEIGH);
    u16* xbf      = (u16*)(ws + WS_XBF);

    k_zero_stats<<<1, 256, 0, stream>>>(stat);
    k_bn_stats<<<480, 256, 0, stream>>>(feat, stat, N);
    k_bn_finalize<<<1, 128, 0, stream>>>(stat, gamma, beta, N);
    k_conv_weights<<<(131072 + 255) / 256, 256, 0, stream>>>(
        Wih, Whh, Wself, Wneigh, wihb, whhb, wselfb, wneighb);
    {
        long groups = (long)N * D / 4;
        int blocks = (int)((groups + 255) / 256);
        k_normalize<<<blocks, 256, 0, stream>>>(feat, stat, xbf, N);
    }
    {
        int tiles  = (N + TILE_M - 1) / TILE_M;
        int blocks = (tiles + WAVES - 1) / WAVES;
        k_eopa_gru<<<blocks, 256, LDS_TOTAL, stream>>>(
            xbf, wihb, wselfb, wneighb, bih, bhh, srcp, mskp, out, N, MAXD);
    }
}
